// CorrelationCost_73933567033724
// MI455X (gfx1250) — matrix operand order repaired and checked by reading, not compiled, not run on hardware
//
#include <hip/hip_runtime.h>

// Problem constants (from the reference)
#define BATCH 8
#define CH    256
#define HH    96
#define WW    192
#define NDD   81     // (2*4+1)^2 displacements
#define LEAK  0.1f

typedef __bf16 v16bf __attribute__((ext_vector_type(16)));
typedef float  v8f   __attribute__((ext_vector_type(8)));

struct B128x2 { uint4 a, b; };   // 32B = one WMMA 16-bit operand fragment

// Pack two floats into one dword of bf16 (RNE via HW convert): low = x0, high = x1.
__device__ __forceinline__ unsigned pk2(float x0, float x1) {
  unsigned short a = __builtin_bit_cast(unsigned short, (__bf16)x0);
  unsigned short b = __builtin_bit_cast(unsigned short, (__bf16)x1);
  return (unsigned)a | ((unsigned)b << 16);
}

__device__ __forceinline__ v8f wmma_bf16(v16bf a, v16bf b, v8f c) {
  // (neg_a, A, neg_b, B, c_mod, C, reuse_a, reuse_b)
  return __builtin_amdgcn_wmma_f32_16x16x32_bf16(false, a, false, b, (short)0, c,
                                                 false, false);
}

// XOR swizzle: dword `d` (0..127) of pixel-row `px` is stored at position
// ((G ^ (px&15))<<2)|j, G=d>>2, j=d&3. Kills DS bank conflicts on b128 reads
// (px stride 512B contributes nothing to the bank index otherwise).
__device__ __forceinline__ int swz(int px, int d) {
  return (((d >> 2) ^ (px & 15)) << 2) | (d & 3);
}

// Workgroup: 4 waves (128 threads). Tile = (b, h0..h0+3, w0..w0+15).
// Wave `wid` computes output row h = h0 + wid for all 81 displacements.
// LDS: ring of 4 feat2 rows, px-major bf16 K-pairs: [slot][px(32)][cpair(128)] u32.
__global__ __launch_bounds__(128) void corr_wmma_kernel(
    const float* __restrict__ f1, const float* __restrict__ f2,
    float* __restrict__ out) {
  __shared__ __align__(16) unsigned int lds[4 * 32 * 128];  // 64 KB

  const int w0 = blockIdx.x * 16;
  const int h0 = blockIdx.y * 4;
  const int b  = blockIdx.z;
  const int t    = threadIdx.x;
  const int lane = t & 31;
  const int wid  = t >> 5;
  const int h    = h0 + wid;

  const size_t hw  = (size_t)HH * WW;
  const size_t chw = (size_t)CH * hw;
  const float* f1b = f1 + (size_t)b * chw;
  const float* f2b = f2 + (size_t)b * chw;

  const int n = lane & 15;  // N column / pixel-in-tile
  const int q = lane >> 4;  // upper/lower K half

  // ---- A fragments: feat1[b, :, h, w0+n] for all 256 channels, bf16 packed ----
  // A 16x32 in this code's contiguous K order (B reads match): lane m = lane&15; VGPR v holds K-pair
  //   kb = kk*32 + q*16 + v*2
  v16bf a[8];
  {
    const float* p = f1b + (size_t)h * WW + (w0 + n);
#pragma unroll
    for (int kk = 0; kk < 8; ++kk) {
      union { unsigned u[8]; v16bf v; } au;
#pragma unroll
      for (int v = 0; v < 8; ++v) {
        const int kb = kk * 32 + q * 16 + v * 2;
        const float x0 = p[(size_t)kb * hw];
        const float x1 = p[(size_t)(kb + 1) * hw];
        au.u[v] = pk2(x0, x1);
      }
      a[kk] = au.v;
    }
  }

  // ---- cooperative stager: feat2 row (h0-4+ridx), px w0-4..w0+27, bf16 to LDS ----
  auto stage_row = [&](int ridx) {
    const int slot = ridx & 3;
    const int hsrc = h0 - 4 + ridx;
    const bool hok = (hsrc >= 0) && (hsrc < HH);
    const int px2  = (t & 15) * 2;       // even pixel pair within 32-px window
    const int wq   = w0 - 4 + px2;       // pair never straddles the [0,W) boundary
    const bool ok  = hok && (wq >= 0) && (wq < WW);
    unsigned* base = &lds[slot * 4096];
    const float* src = f2b + (size_t)hsrc * WW + wq;
    const int cp0 = t >> 4;              // 0..7
#pragma unroll
    for (int pI = 0; pI < 16; ++pI) {
      const int cpair = cp0 + pI * 8;    // channel pair 0..127
      float x00 = 0.f, x01 = 0.f, x10 = 0.f, x11 = 0.f;
      if (ok) {
        const float* s0 = src + (size_t)(cpair * 2) * hw;
        x00 = s0[0]; x01 = s0[1];        // channel c, px wq / wq+1 (contiguous)
        const float* s1 = s0 + hw;       // channel c+1
        x10 = s1[0]; x11 = s1[1];
      }
      base[(px2 + 0) * 128 + swz(px2 + 0, cpair)] = pk2(x00, x10);
      base[(px2 + 1) * 128 + swz(px2 + 1, cpair)] = pk2(x01, x11);
    }
  };

  // Pre-stage rows ridx 0..3 (hsrc = h0-4 .. h0-1 for dy=0)
#pragma unroll
  for (int r0 = 0; r0 < 4; ++r0) stage_row(r0);
  __syncthreads();

  const float invC = 1.0f / (float)CH;

  for (int dy = 0; dy < 9; ++dy) {
    // This wave's source row: hsrc = h + dy - 4 -> ridx = wid + dy
    const int slot = (wid + dy) & 3;

    // Per-lane fragment rows: B0 -> px n (w0-4..w0+11), B1 -> px 16+n (w0+12..w0+27)
    const uint4* row0 = (const uint4*)&lds[slot * 4096 + n * 128];
    const uint4* row1 = (const uint4*)&lds[slot * 4096 + (16 + n) * 128];

    v8f c0 = {};
    v8f c1 = {};
#pragma unroll
    for (int kk = 0; kk < 8; ++kk) {
      // Fragment = dwords [kk*16+q*8 .. +7] of the lane's px-row: granules 2c,2c+1.
      const int c = kk * 2 + q;             // chunk 0..15
      const int g0i = (2 * c) ^ n;          // swizzled granule positions
      const int g1i = (2 * c + 1) ^ n;
      B128x2 b0, b1;
      b0.a = row0[g0i]; b0.b = row0[g1i];   // two ds_load_b128, conflict-free
      b1.a = row1[g0i]; b1.b = row1[g1i];
      c0 = wmma_bf16(a[kk], __builtin_bit_cast(v16bf, b0), c0);
      c1 = wmma_bf16(a[kk], __builtin_bit_cast(v16bf, b1), c1);
    }

    // Band extraction: C[m][n] = corr(pixel w0+m, dx_idx = n-m (+16 for B1)).
    // mean over C, leaky relu, store.
#pragma unroll
    for (int r = 0; r < 8; ++r) {
      const int m = r + q * 8;
      float v0 = c0[r] * invC;
      v0 = (v0 > 0.f) ? v0 : v0 * LEAK;
      const int dxi0 = n - m;
      if (dxi0 >= 0 && dxi0 <= 8)
        out[(((size_t)b * NDD + dy * 9 + dxi0) * HH + h) * WW + (w0 + m)] = v0;

      float v1 = c1[r] * invC;
      v1 = (v1 > 0.f) ? v1 : v1 * LEAK;
      const int dxi1 = n + 16 - m;
      if (dxi1 >= 0 && dxi1 <= 8)
        out[(((size_t)b * NDD + dy * 9 + dxi1) * HH + h) * WW + (w0 + m)] = v1;
    }

    __syncthreads();
    if (dy < 8) {            // slide the ring: bring in row ridx = dy+4
      stage_row(dy + 4);
      __syncthreads();
    }
  }
}

extern "C" void kernel_launch(void* const* d_in, const int* in_sizes, int n_in,
                              void* d_out, int out_size, void* d_ws, size_t ws_size,
                              hipStream_t stream) {
  (void)in_sizes; (void)n_in; (void)out_size; (void)d_ws; (void)ws_size;
  const float* f1 = (const float*)d_in[0];
  const float* f2 = (const float*)d_in[1];
  float* out = (float*)d_out;
  dim3 grid(WW / 16, HH / 4, BATCH);   // 12 x 24 x 8 workgroups
  dim3 block(128);                     // 4 wave32s
  hipLaunchKernelGGL(corr_wmma_kernel, grid, block, 0, stream, f1, f2, out);
}
